// ModSelfAttention_28930899706119
// MI455X (gfx1250) — matrix operand order repaired and checked by reading, not compiled, not run on hardware
//
#include <hip/hip_runtime.h>

typedef __attribute__((ext_vector_type(16))) __bf16 v16bf;
typedef __attribute__((ext_vector_type(8)))  float  v8f;
typedef unsigned int u32;
typedef __attribute__((ext_vector_type(4))) u32 u32x4;
typedef __attribute__((ext_vector_type(8))) u32 u32x8;

#define D_MODEL 1024
#define N_HEAD  16
#define SEQ     38
#define BATCH   2048
#define M_TOTAL (BATCH*SEQ)       /* 77824 */
#define BH      (BATCH*N_HEAD)    /* 32768 */

struct alignas(16) U4 { unsigned int v[4]; };
struct alignas(16) F4 { float v[4]; };
union Frag { U4 u[2]; v16bf b; };

__device__ __forceinline__ unsigned short f2bf(float f){
  unsigned int u = __float_as_uint(f);
  u += 0x7FFFu + ((u >> 16) & 1u);     // round-to-nearest-even
  return (unsigned short)(u >> 16);
}

__device__ __forceinline__ v8f wmma_bf16(const Frag& a, const Frag& b, v8f c){
  return __builtin_amdgcn_wmma_f32_16x16x32_bf16(false, a.b, false, b.b,
                                                 (short)0, c, false, false);
}

// --- CDNA5 async copy: per-lane 16B global -> LDS, tracked by ASYNCcnt -----
__device__ __forceinline__ void async_g2l_b128(const void* gp, void* lp){
  asm volatile("global_load_async_to_lds_b128 %0, %1, off"
               :: "v"((u32)(unsigned long long)lp), "v"(gp)
               : "memory");
}
__device__ __forceinline__ void wait_async0(){
  asm volatile("s_wait_asynccnt 0x0" ::: "memory");
}

// --- CDNA5 Tensor Data Mover: 2D tile (tile_k elems x tile_rows), bf16,
// row stride in elems; LDS padded 16 DWORDs data + 4 DWORDs pad per row
// (=> LDS row pitch of 40 bf16, matching the conflict-free sA/sB layout). ---
__device__ __forceinline__ void tdm_load_tile_2d(const void* gaddr, void* lds,
    u32 tile_k_elems, u32 tile_rows, u32 row_stride_elems){
  const unsigned long long ga = (unsigned long long)(size_t)gaddr;
  u32x4 g0;
  g0[0] = 1u;                                              // count=1, load
  g0[1] = (u32)(unsigned long long)(size_t)lds;            // lds_addr (bytes)
  g0[2] = (u32)ga;                                         // global_addr lo
  g0[3] = ((u32)(ga >> 32) & 0x01FFFFFFu) | (2u << 30);    // addr hi | type=2
  const u32 td0 = row_stride_elems;                        // tensor_dim0
  const u32 td1 = 0x00100000u;                             // tensor_dim1 (big)
  u32x8 g1;
  g1[0] = (1u << 16)            // data_size = 2 bytes
        | (1u << 20)            // pad_enable
        | (3u << 22)            // pad_interval: 16 DWORDs (64B)
        | (3u << 25);           // pad_amount : 4 DWORDs (16B)
  g1[1] = (td0 & 0xFFFFu) << 16;                           // tensor_dim0 lo16
  g1[2] = ((td0 >> 16) & 0xFFFFu) | ((td1 & 0xFFFFu) << 16);
  g1[3] = ((td1 >> 16) & 0xFFFFu) | ((tile_k_elems & 0xFFFFu) << 16); // tile_dim0
  g1[4] = (tile_rows & 0xFFFFu);                           // tile_dim1
  g1[5] = row_stride_elems;                                // tensor_dim0_stride lo
  g1[6] = 0u;
  g1[7] = 0u;
  asm volatile("tensor_load_to_lds %0, %1" :: "s"(g0), "s"(g1) : "memory");
}

// ---------------------------------------------------------------------------
// Kernel 1: pre-LN (mean/var over 1024) -> bf16 xn rows
// ---------------------------------------------------------------------------
__global__ __launch_bounds__(256) void ln_kernel(
    const float* __restrict__ x, const float* __restrict__ g,
    const float* __restrict__ be, unsigned short* __restrict__ xn){
  const int row = blockIdx.x;
  const float* xr = x + (size_t)row * D_MODEL;
  const int t = threadIdx.x, lane = t & 31, w = t >> 5;
  F4 xv = *(const F4*)(xr + t * 4);
  float s1 = 0.f, s2 = 0.f;
#pragma unroll
  for(int i = 0; i < 4; i++){ s1 += xv.v[i]; s2 += xv.v[i] * xv.v[i]; }
#pragma unroll
  for(int o = 16; o > 0; o >>= 1){
    s1 += __shfl_xor(s1, o, 32);
    s2 += __shfl_xor(s2, o, 32);
  }
  __shared__ float r1[8], r2[8];
  if(lane == 0){ r1[w] = s1; r2[w] = s2; }
  __syncthreads();
  float a1 = 0.f, a2 = 0.f;
#pragma unroll
  for(int i = 0; i < 8; i++){ a1 += r1[i]; a2 += r2[i]; }
  const float mean = a1 * (1.f / 1024.f);
  const float var  = a2 * (1.f / 1024.f) - mean * mean;
  const float rstd = rsqrtf(var + 1e-6f);
  F4 gv = *(const F4*)(g  + t * 4);
  F4 bv = *(const F4*)(be + t * 4);
  unsigned long long p = 0;
#pragma unroll
  for(int i = 0; i < 4; i++){
    unsigned long long h = f2bf((xv.v[i] - mean) * rstd * gv.v[i] + bv.v[i]);
    p |= h << (16 * i);
  }
  *(unsigned long long*)(xn + (size_t)row * D_MODEL + t * 4) = p;
}

// ---------------------------------------------------------------------------
// Kernel 2: convert weights to bf16, TRANSPOSED to [out(N)][in(K)].
// 9 matrices (q/k/v x widx 0..2) then fc_w.
// ---------------------------------------------------------------------------
__global__ __launch_bounds__(256) void wconv_kernel(
    const float* __restrict__ Wq, const float* __restrict__ Wk,
    const float* __restrict__ Wv, const float* __restrict__ fcw,
    unsigned short* __restrict__ Wt, unsigned short* __restrict__ fcWt){
  const size_t idx = (size_t)blockIdx.x * 256 + threadIdx.x;  // 10M total
  const int mat = (int)(idx >> 20);
  const int r = (int)((idx >> 10) & 1023);  // out (row of transposed)
  const int c = (int)(idx & 1023);          // in
  const float* src;
  unsigned short* dst;
  if(mat < 9){
    const int z = mat / 3, wi = mat % 3;
    src = (z == 0 ? Wq : (z == 1 ? Wk : Wv)) + ((size_t)wi << 20);
    dst = Wt + ((size_t)mat << 20);
  } else { src = fcw; dst = fcWt; }
  dst[((size_t)r << 10) | c] = f2bf(src[((size_t)c << 10) | r]);
}

// ---------------------------------------------------------------------------
// Kernel 3: segment-grouped QKV GEMM. 128x128 tile per block, 8 waves.
// A tile: per-lane async-to-LDS (rows are segment-permuted).
// B tile: one TDM descriptor issued by wave 0 (regular strided weights).
// ---------------------------------------------------------------------------
__global__ __launch_bounds__(256) void qkv_gemm(
    const unsigned short* __restrict__ xn, const unsigned short* __restrict__ Wt,
    unsigned short* __restrict__ qb, unsigned short* __restrict__ kb,
    unsigned short* __restrict__ vT){
  __shared__ __align__(16) unsigned short sA[128 * 40];
  __shared__ __align__(16) unsigned short sB[128 * 40];
  const int bx = blockIdx.x, nB = blockIdx.y, z = blockIdx.z;
  int g, gr0;
  if(bx < 512){ g = 0; gr0 = bx << 7; }
  else if(bx < 560){ g = 1; gr0 = (bx - 512) << 7; }
  else { g = 2; gr0 = (bx - 560) << 7; }
  const unsigned short* Bm = Wt + ((size_t)(z * 3 + g) << 20);
  const int tid = threadIdx.x;
  const int lane = tid & 31, wv = tid >> 5;
  const int wm = wv >> 2, wn = wv & 3;
  const int l15 = lane & 15, hi = lane >> 4;
  const int r0 = tid >> 2, cc = tid & 3;

  auto maprow = [g](int gr)->int {
    int b, l;
    if(g == 0){ b = gr >> 5; int j = gr & 31; l = (j < 29) ? j : (j + 3); }
    else { b = gr / 3; int j = gr - b * 3; l = ((g == 1) ? 29 : 35) + j; }
    return b * SEQ + l;
  };
  const size_t arow0 = (size_t)maprow(gr0 + r0)       << 10;
  const size_t arow1 = (size_t)maprow(gr0 + r0 + 64)  << 10;
  const unsigned short* Brow = Bm + ((size_t)(nB * 128) << 10);

  v8f acc[4][2];
#pragma unroll
  for(int i = 0; i < 4; i++)
#pragma unroll
    for(int j = 0; j < 2; j++)
#pragma unroll
      for(int r = 0; r < 8; r++) acc[i][j][r] = 0.f;

  for(int kk = 0; kk < 1024; kk += 32){
    __syncthreads();
    async_g2l_b128(&xn[arow0 + kk + cc * 8], &sA[r0 * 40 + cc * 8]);
    async_g2l_b128(&xn[arow1 + kk + cc * 8], &sA[(r0 + 64) * 40 + cc * 8]);
    if(wv == 0)
      tdm_load_tile_2d(Brow + kk, sB, 32u, 128u, 1024u);
    wait_async0();
    __builtin_amdgcn_s_wait_tensorcnt(0);
    __syncthreads();
    Frag a[4], b[2];
#pragma unroll
    for(int j = 0; j < 2; j++){
      const int col = wn * 32 + j * 16 + l15;
      const int k0 = hi ? 8 : 0;
      b[j].u[0] = *(const U4*)&sB[col * 40 + k0];
      b[j].u[1] = *(const U4*)&sB[col * 40 + k0 + 16];
    }
#pragma unroll
    for(int i = 0; i < 4; i++){
      const int row = wm * 64 + i * 16 + l15;
      const int k0 = hi ? 8 : 0;
      a[i].u[0] = *(const U4*)&sA[row * 40 + k0];
      a[i].u[1] = *(const U4*)&sA[row * 40 + k0 + 16];
    }
#pragma unroll
    for(int i = 0; i < 4; i++)
#pragma unroll
      for(int j = 0; j < 2; j++)
        acc[i][j] = wmma_bf16(a[i], b[j], acc[i][j]);
  }

#pragma unroll
  for(int i = 0; i < 4; i++){
#pragma unroll
    for(int j = 0; j < 2; j++){
      const int n = nB * 128 + wn * 32 + j * 16 + l15;
      const int h = n >> 6, d = n & 63;
#pragma unroll
      for(int r = 0; r < 8; r++){
        const int gr = gr0 + wm * 64 + i * 16 + hi * 8 + r;
        int b_, l;
        if(g == 0){ b_ = gr >> 5; int jj = gr & 31; l = (jj < 29) ? jj : (jj + 3); }
        else { b_ = gr / 3; int jj = gr - b_ * 3; l = ((g == 1) ? 29 : 35) + jj; }
        const unsigned short val = f2bf(acc[i][j][r]);
        const size_t bh = (size_t)(b_ * N_HEAD + h);
        if(z == 0)      qb[(bh * SEQ + l) * 64 + d] = val;
        else if(z == 1) kb[(bh * SEQ + l) * 64 + d] = val;
        else            vT[(bh * 64 + d) * 48 + l]  = val;   // transposed V
      }
    }
  }
}

// ---------------------------------------------------------------------------
// Kernel 4: attention per (b,h). One wave. scores(48x48 padded, 18 wmma)
// -> softmax in LDS -> fp32 probs to d_out -> attn@V (24 wmma) -> bf16 head out.
// ---------------------------------------------------------------------------
__global__ __launch_bounds__(32) void attn_kernel(
    const unsigned short* __restrict__ qb, const unsigned short* __restrict__ kb,
    const unsigned short* __restrict__ vT, unsigned short* __restrict__ ao,
    float* __restrict__ attn){
  __shared__ __align__(16) float          s[48 * 49];
  __shared__ __align__(16) unsigned short pb[48 * 72];
  const int bh = blockIdx.x;
  const int lane = threadIdx.x, l15 = lane & 15, hi = lane >> 4;
  const unsigned short* q = qb + (size_t)bh * SEQ * 64;
  const unsigned short* k = kb + (size_t)bh * SEQ * 64;
  const unsigned short* v = vT + (size_t)bh * 64 * 48;

  // scores = q k^T / 8
#pragma unroll
  for(int tm = 0; tm < 3; tm++){
#pragma unroll
    for(int tn = 0; tn < 3; tn++){
      v8f acc;
#pragma unroll
      for(int r = 0; r < 8; r++) acc[r] = 0.f;
#pragma unroll
      for(int kc = 0; kc < 2; kc++){
        Frag a, b;
        const int ar = tm * 16 + l15, ak = kc * 32 + hi * 8;
        a.u[0] = *(const U4*)&q[ar * 64 + ak];
        a.u[1] = *(const U4*)&q[ar * 64 + ak + 16];
        const int bc = tn * 16 + l15, bk = kc * 32 + hi * 8;
        b.u[0] = *(const U4*)&k[bc * 64 + bk];
        b.u[1] = *(const U4*)&k[bc * 64 + bk + 16];
        acc = wmma_bf16(a, b, acc);
      }
#pragma unroll
      for(int r = 0; r < 8; r++)
        s[(tm * 16 + hi * 8 + r) * 49 + tn * 16 + l15] = acc[r] * 0.125f;
    }
  }
  __syncthreads();

  // softmax rows (rows >= SEQ are padding; masked at stores)
  for(int r = lane; r < 48; r += 32){
    float mx = -1e30f;
    for(int c = 0; c < SEQ; c++) mx = fmaxf(mx, s[r * 49 + c]);
    float sum = 0.f;
    for(int c = 0; c < SEQ; c++){
      const float e = __expf(s[r * 49 + c] - mx);
      s[r * 49 + c] = e; sum += e;
    }
    const float inv = 1.f / sum;
    for(int c = 0; c < SEQ; c++){
      const float p = s[r * 49 + c] * inv;
      s[r * 49 + c] = p;
      pb[r * 72 + c] = f2bf(p);
    }
    for(int c = SEQ; c < 64; c++) pb[r * 72 + c] = 0;  // zero K padding
  }
  __syncthreads();

  // write fp32 attention probabilities
  float* aout = attn + (size_t)bh * SEQ * SEQ;
  for(int idx = lane; idx < SEQ * SEQ; idx += 32){
    const int qi = idx / SEQ, ki = idx - qi * SEQ;
    aout[idx] = s[qi * 49 + ki];
  }

  // out = probs @ V  (A from LDS pb, B = vT rows)
  const int b_ = bh >> 4, h = bh & 15;
#pragma unroll
  for(int tm = 0; tm < 3; tm++){
#pragma unroll
    for(int tn = 0; tn < 4; tn++){
      v8f acc;
#pragma unroll
      for(int r = 0; r < 8; r++) acc[r] = 0.f;
#pragma unroll
      for(int kc = 0; kc < 2; kc++){
        Frag a, b;
        const int ar = tm * 16 + l15, ak = kc * 32 + hi * 8;
        a.u[0] = *(const U4*)&pb[ar * 72 + ak];
        a.u[1] = *(const U4*)&pb[ar * 72 + ak + 16];
        const int bc = tn * 16 + l15, bk = kc * 32 + hi * 8;
        b.u[0] = *(const U4*)&v[bc * 48 + bk];
        b.u[1] = *(const U4*)&v[bc * 48 + bk + 16];
        acc = wmma_bf16(a, b, acc);
      }
#pragma unroll
      for(int r = 0; r < 8; r++){
        const int m = tm * 16 + hi * 8 + r;
        if(m < SEQ){
          const int d = tn * 16 + l15;
          ao[((size_t)(b_ * SEQ + m) << 10) + h * 64 + d] = f2bf(acc[r]);
        }
      }
    }
  }
}

// ---------------------------------------------------------------------------
// Kernel 5: fc GEMM + bias + residual -> fp32 output.
// Both A and B tiles are regular strided: wave 0 TDMs A, wave 1 TDMs B.
// ---------------------------------------------------------------------------
__global__ __launch_bounds__(256) void fc_gemm(
    const unsigned short* __restrict__ ao, const unsigned short* __restrict__ Wt,
    const float* __restrict__ x, const float* __restrict__ fcb,
    float* __restrict__ out){
  __shared__ __align__(16) unsigned short sA[128 * 40];
  __shared__ __align__(16) unsigned short sB[128 * 40];
  const int bx = blockIdx.x, nB = blockIdx.y;
  const int tid = threadIdx.x;
  const int lane = tid & 31, wv = tid >> 5;
  const int wm = wv >> 2, wn = wv & 3;
  const int l15 = lane & 15, hi = lane >> 4;
  const unsigned short* Arow = ao + ((size_t)(bx * 128) << 10);
  const unsigned short* Brow = Wt + ((size_t)(nB * 128) << 10);

  v8f acc[4][2];
#pragma unroll
  for(int i = 0; i < 4; i++)
#pragma unroll
    for(int j = 0; j < 2; j++)
#pragma unroll
      for(int r = 0; r < 8; r++) acc[i][j][r] = 0.f;

  for(int kk = 0; kk < 1024; kk += 32){
    __syncthreads();
    if(wv == 0)      tdm_load_tile_2d(Arow + kk, sA, 32u, 128u, 1024u);
    else if(wv == 1) tdm_load_tile_2d(Brow + kk, sB, 32u, 128u, 1024u);
    __builtin_amdgcn_s_wait_tensorcnt(0);
    __syncthreads();
    Frag a[4], b[2];
#pragma unroll
    for(int j = 0; j < 2; j++){
      const int col = wn * 32 + j * 16 + l15;
      const int k0 = hi ? 8 : 0;
      b[j].u[0] = *(const U4*)&sB[col * 40 + k0];
      b[j].u[1] = *(const U4*)&sB[col * 40 + k0 + 16];
    }
#pragma unroll
    for(int i = 0; i < 4; i++){
      const int row = wm * 64 + i * 16 + l15;
      const int k0 = hi ? 8 : 0;
      a[i].u[0] = *(const U4*)&sA[row * 40 + k0];
      a[i].u[1] = *(const U4*)&sA[row * 40 + k0 + 16];
    }
#pragma unroll
    for(int i = 0; i < 4; i++)
#pragma unroll
      for(int j = 0; j < 2; j++)
        acc[i][j] = wmma_bf16(a[i], b[j], acc[i][j]);
  }

#pragma unroll
  for(int i = 0; i < 4; i++){
#pragma unroll
    for(int j = 0; j < 2; j++){
      const int n = nB * 128 + wn * 32 + j * 16 + l15;
      const float bias = fcb[n];
#pragma unroll
      for(int r = 0; r < 8; r++){
        const int m = bx * 128 + wm * 64 + i * 16 + hi * 8 + r;
        const size_t o = ((size_t)m << 10) + n;
        out[o] = acc[i][j][r] + bias + x[o];
      }
    }
  }
}

// ---------------------------------------------------------------------------
extern "C" void kernel_launch(void* const* d_in, const int* in_sizes, int n_in,
                              void* d_out, int out_size, void* d_ws, size_t ws_size,
                              hipStream_t stream){
  const float* x   = (const float*)d_in[0];
  const float* Wq  = (const float*)d_in[1];
  const float* Wk  = (const float*)d_in[2];
  const float* Wv  = (const float*)d_in[3];
  const float* fcw = (const float*)d_in[4];
  const float* fcb = (const float*)d_in[5];
  const float* lng = (const float*)d_in[6];
  const float* lnb = (const float*)d_in[7];
  float* out  = (float*)d_out;
  float* attn = out + (size_t)M_TOTAL * D_MODEL;

  char* ws = (char*)d_ws;
  const size_t SZ_XN = (size_t)M_TOTAL * D_MODEL * 2;   // also reused as attn-out bf16
  const size_t SZ_QK = (size_t)BH * SEQ * 64 * 2;
  const size_t SZ_VT = (size_t)BH * 64 * 48 * 2;
  unsigned short* xn   = (unsigned short*)ws;
  unsigned short* qb   = (unsigned short*)(ws + SZ_XN);
  unsigned short* kb   = (unsigned short*)(ws + SZ_XN + SZ_QK);
  unsigned short* vT   = (unsigned short*)(ws + SZ_XN + 2 * SZ_QK);
  unsigned short* Wt   = (unsigned short*)(ws + SZ_XN + 2 * SZ_QK + SZ_VT);
  unsigned short* fcWt = Wt + ((size_t)9 << 20);

  ln_kernel<<<M_TOTAL, 256, 0, stream>>>(x, lng, lnb, xn);
  wconv_kernel<<<(10u << 20) / 256, 256, 0, stream>>>(Wq, Wk, Wv, fcw, Wt, fcWt);
  qkv_gemm<<<dim3(608, 8, 3), 256, 0, stream>>>(xn, Wt, qb, kb, vT);
  attn_kernel<<<BH, 32, 0, stream>>>(qb, kb, vT, xn /*attn-out reuses xn*/, attn);
  fc_gemm<<<dim3(608, 8), 256, 0, stream>>>(xn, fcWt, x, fcb, out);
}
